// SparseMLP_42966852829514
// MI455X (gfx1250) — compile-verified, matrix operand order checked by reading, not run on hardware
//
#include <hip/hip_runtime.h>
#include <hip/hip_bf16.h>
#include <math.h>

// ---------------------------------------------------------------------------
// Grouped MoE MLP:  out[e] = gelu_tanh(x[e] @ w1[e]^T) @ w2[e]
//   E=8, T=2048, H=1024, F=4096.  bf16 WMMA, f32 accumulate.
// Software-pipelined: global loads for chunk k+1 in flight during WMMA of
// chunk k; double-buffered LDS; one barrier per k-chunk. GEMM2's A tile
// (bf16 workspace) uses GLOBAL_LOAD_ASYNC_TO_LDS_B128 when available.
// ---------------------------------------------------------------------------

#define NE 8
#define NT 2048
#define NH 1024
#define NF 4096

typedef __attribute__((ext_vector_type(16))) __bf16 v16bf;
typedef __attribute__((ext_vector_type(8)))  __bf16 v8bf;
typedef __attribute__((ext_vector_type(4)))  __bf16 v4bf;
typedef __attribute__((ext_vector_type(8)))  float  v8f;
typedef __attribute__((ext_vector_type(4)))  int    v4i;

#define BM 128
#define BN 128
#define BK 64
#define ATILE (BM * BK)
#define BTILE (BN * BK)

#if defined(__AMDGCN__) && __has_builtin(__builtin_amdgcn_global_load_async_to_lds_b128)
#define HAVE_ASYNC_LDS 1
typedef __attribute__((address_space(1))) v4i g_v4i;   // global int4
typedef __attribute__((address_space(3))) v4i l_v4i;   // LDS int4
#else
#define HAVE_ASYNC_LDS 0
#endif

__device__ __forceinline__ void wait_async0() {
#if HAVE_ASYNC_LDS
#if __has_builtin(__builtin_amdgcn_s_wait_asynccnt)
    __builtin_amdgcn_s_wait_asynccnt(0);
#else
    asm volatile("s_wait_asynccnt 0x0" ::: "memory");
#endif
#endif
}

// WMMA A/B fragment from an LDS tile stored K-contiguous (stride BK).
// CDNA5 16-bit operand layout: lane holds K = kb..kb+7 and kb+16..kb+23,
// kb = kstep*32 + (lane>=16 ? 8 : 0).
__device__ __forceinline__ v16bf load_frag(const __bf16* lds, int row, int kb) {
    const __bf16* p = lds + row * BK + kb;
    v8bf lo = *(const v8bf*)(p);        // ds_load_b128
    v8bf hi = *(const v8bf*)(p + 16);   // ds_load_b128
    v16bf f;
#pragma unroll
    for (int i = 0; i < 8; ++i) { f[i] = lo[i]; f[i + 8] = hi[i]; }
    return f;
}

// 8 WMMA tiles (wave computes 64x32) over one 64-deep K chunk.
__device__ __forceinline__ void wmma_chunk(const __bf16* la, const __bf16* lb,
                                           v8f acc[4][2], int lane, int wm, int wn) {
#pragma unroll
    for (int ks = 0; ks < 2; ++ks) {
        const int kb = ks * 32 + ((lane >> 4) << 3);
        v16bf af[4], bfr[2];
#pragma unroll
        for (int mi = 0; mi < 4; ++mi)
            af[mi] = load_frag(la, wm + mi * 16 + (lane & 15), kb);
#pragma unroll
        for (int ni = 0; ni < 2; ++ni)
            bfr[ni] = load_frag(lb, wn + ni * 16 + (lane & 15), kb);
#pragma unroll
        for (int mi = 0; mi < 4; ++mi)
#pragma unroll
            for (int ni = 0; ni < 2; ++ni)
                acc[mi][ni] = __builtin_amdgcn_wmma_f32_16x16x32_bf16(
                    false, af[mi], false, bfr[ni], (short)0, acc[mi][ni],
                    false, false);
    }
}

// ---- f32 tile staging (128 rows x 64 K), two-phase: loads then cvt+store ----
__device__ __forceinline__ void gload_f32(const float* __restrict__ src,
                                          int stride, float4* r, int tid) {
#pragma unroll
    for (int i = 0; i < 8; ++i) {
        int idx = tid + i * 256;            // 2048 float4 units
        int row = idx >> 4;
        int c4  = (idx & 15) << 2;
        r[i] = *(const float4*)(src + (size_t)row * stride + c4);
    }
}

__device__ __forceinline__ void cvt_store_f32(const float4* r, __bf16* dst, int tid) {
#pragma unroll
    for (int i = 0; i < 8; ++i) {
        int idx = tid + i * 256;
        int row = idx >> 4;
        int c4  = (idx & 15) << 2;
        v4bf b;
        b[0] = (__bf16)r[i].x; b[1] = (__bf16)r[i].y;
        b[2] = (__bf16)r[i].z; b[3] = (__bf16)r[i].w;
        *(v4bf*)(dst + row * BK + c4) = b;   // ds_store_b64
    }
}

__device__ __forceinline__ float gelu_tanh(float x) {
    const float k0 = 0.7978845608028654f;   // sqrt(2/pi)
    const float k1 = 0.044715f;
    float u = k0 * (x + k1 * x * x * x);
    float e = __expf(u + u);                 // v_exp_f32
    return 0.5f * x * (1.0f + (e - 1.0f) / (e + 1.0f));
}

// ---------------------------------------------------------------------------
// Kernel 1: a[e] = gelu(x[e] @ w1[e]^T)   M=T, N=F, K=H  (NT: both K-contig)
// ---------------------------------------------------------------------------
__global__ __launch_bounds__(256)
void moe_gemm1_gelu(const float* __restrict__ x, const float* __restrict__ w1,
                    __bf16* __restrict__ a) {
    __shared__ __bf16 lds_a[2 * ATILE];
    __shared__ __bf16 lds_b[2 * BTILE];

    const int tid  = threadIdx.x;
    const int lane = tid & 31;
    const int w    = tid >> 5;
    const int wm   = (w & 1) * 64;
    const int wn   = (w >> 1) * 32;

    const int e  = blockIdx.z;
    const int m0 = blockIdx.y * BM;
    const int n0 = blockIdx.x * BN;

    const float* xa = x  + (size_t)e * NT * NH + (size_t)m0 * NH;
    const float* wb = w1 + (size_t)e * NF * NH + (size_t)n0 * NH;

    v8f acc[4][2];
#pragma unroll
    for (int mi = 0; mi < 4; ++mi)
#pragma unroll
        for (int ni = 0; ni < 2; ++ni) acc[mi][ni] = (v8f)(0.0f);

    float4 ra[8], rb[8];
    gload_f32(xa, NH, ra, tid);
    gload_f32(wb, NH, rb, tid);

    int buf = 0;
    for (int k0 = 0; k0 < NH; k0 += BK, buf ^= 1) {
        cvt_store_f32(ra, lds_a + buf * ATILE, tid);
        cvt_store_f32(rb, lds_b + buf * BTILE, tid);
        if (k0 + BK < NH) {                  // next chunk in flight during WMMA
            gload_f32(xa + k0 + BK, NH, ra, tid);
            gload_f32(wb + k0 + BK, NH, rb, tid);
        }
        __syncthreads();
        wmma_chunk(lds_a + buf * ATILE, lds_b + buf * BTILE, acc, lane, wm, wn);
    }

    __bf16* ao = a + (size_t)e * NT * NF;
#pragma unroll
    for (int mi = 0; mi < 4; ++mi)
#pragma unroll
        for (int ni = 0; ni < 2; ++ni) {
            const int col  = n0 + wn + ni * 16 + (lane & 15);
            const int rbse = m0 + wm + mi * 16 + ((lane >> 4) << 3);
#pragma unroll
            for (int r = 0; r < 8; ++r)
                ao[(size_t)(rbse + r) * NF + col] = (__bf16)gelu_tanh(acc[mi][ni][r]);
        }
}

// ---------------------------------------------------------------------------
// Kernel 2: out[e] = a[e] @ w2[e]   M=T, N=H, K=F  (NN: B transposed in stage)
// A tile is bf16 already -> async global->LDS DMA when available.
// ---------------------------------------------------------------------------
__global__ __launch_bounds__(256)
void moe_gemm2(const __bf16* __restrict__ a, const float* __restrict__ w2,
               float* __restrict__ out) {
    __shared__ __bf16 lds_a[2 * ATILE];
    __shared__ __bf16 lds_b[2 * BTILE];

    const int tid  = threadIdx.x;
    const int lane = tid & 31;
    const int w    = tid >> 5;
    const int wm   = (w & 1) * 64;
    const int wn   = (w >> 1) * 32;

    const int e  = blockIdx.z;
    const int m0 = blockIdx.y * BM;
    const int n0 = blockIdx.x * BN;

    const __bf16* aa = a  + (size_t)e * NT * NF + (size_t)m0 * NF;
    const float*  wb = w2 + (size_t)e * NF * NH + n0;

    v8f acc[4][2];
#pragma unroll
    for (int mi = 0; mi < 4; ++mi)
#pragma unroll
        for (int ni = 0; ni < 2; ++ni) acc[mi][ni] = (v8f)(0.0f);

#if !HAVE_ASYNC_LDS
    v8bf rav[4];    // A fallback staging: bf16, 4 x 16B per thread
#endif
    float4 rbv[8];  // B: f32, 8 x 16B per thread (rows of w2, N-contiguous)

#if HAVE_ASYNC_LDS
    // Async DMA: each thread copies 4 x 16B of the bf16 A tile straight to LDS.
    // Builtin signature (from clang diagnostic): (as1 int4*, as3 int4*, imm, imm).
#define G2_A_STAGE(K0, BUF)                                                   \
    _Pragma("unroll")                                                         \
    for (int i = 0; i < 4; ++i) {                                             \
        int idx = tid + i * 256; int row = idx >> 3; int c8 = (idx & 7) << 3; \
        __builtin_amdgcn_global_load_async_to_lds_b128(                       \
            (g_v4i*)(aa + (size_t)row * NF + (K0) + c8),                      \
            (l_v4i*)(lds_a + (BUF) * ATILE + row * BK + c8), 0, 0);           \
    }
#else
#define G2_LOAD_A(K0)                                                         \
    _Pragma("unroll")                                                         \
    for (int i = 0; i < 4; ++i) {                                             \
        int idx = tid + i * 256; int row = idx >> 3; int c8 = (idx & 7) << 3; \
        rav[i] = *(const v8bf*)(aa + (size_t)row * NF + (K0) + c8);           \
    }
#define G2_STORE_A(BUF)                                                       \
    _Pragma("unroll")                                                         \
    for (int i = 0; i < 4; ++i) {                                             \
        int idx = tid + i * 256; int row = idx >> 3; int c8 = (idx & 7) << 3; \
        *(v8bf*)(lds_a + (BUF) * ATILE + row * BK + c8) = rav[i];             \
    }
#endif
#define G2_LOAD_B(K0)                                                         \
    _Pragma("unroll")                                                         \
    for (int i = 0; i < 8; ++i) {                                             \
        int idx = tid + i * 256; int k = idx >> 5; int c4 = (idx & 31) << 2;  \
        rbv[i] = *(const float4*)(wb + (size_t)((K0) + k) * NH + c4);         \
    }
#define G2_STORE_B(BUF)                                                       \
    _Pragma("unroll")                                                         \
    for (int i = 0; i < 8; ++i) {                                             \
        int idx = tid + i * 256; int k = idx >> 5; int c4 = (idx & 31) << 2;  \
        __bf16* d = lds_b + (BUF) * BTILE + k;                                \
        d[(size_t)(c4 + 0) * BK] = (__bf16)rbv[i].x;                          \
        d[(size_t)(c4 + 1) * BK] = (__bf16)rbv[i].y;                          \
        d[(size_t)(c4 + 2) * BK] = (__bf16)rbv[i].z;                          \
        d[(size_t)(c4 + 3) * BK] = (__bf16)rbv[i].w;                          \
    }

#if HAVE_ASYNC_LDS
    G2_A_STAGE(0, 0)
#else
    G2_LOAD_A(0)
#endif
    G2_LOAD_B(0)

    int buf = 0;
    for (int k0 = 0; k0 < NF; k0 += BK, buf ^= 1) {
#if !HAVE_ASYNC_LDS
        G2_STORE_A(buf)
#endif
        G2_STORE_B(buf)
        if (k0 + BK < NF) {
#if !HAVE_ASYNC_LDS
            G2_LOAD_A(k0 + BK)
#endif
            G2_LOAD_B(k0 + BK)
        }
        wait_async0();                 // A tile for chunk k resident in LDS
        __syncthreads();
#if HAVE_ASYNC_LDS
        // Issue after the barrier: buffer buf^1 was last read at compute(k-1),
        // which every wave finished before this barrier. DMA overlaps WMMA.
        if (k0 + BK < NF) { G2_A_STAGE(k0 + BK, buf ^ 1) }
#endif
        wmma_chunk(lds_a + buf * ATILE, lds_b + buf * BTILE, acc, lane, wm, wn);
    }

    float* oo = out + (size_t)e * NT * NH;
#pragma unroll
    for (int mi = 0; mi < 4; ++mi)
#pragma unroll
        for (int ni = 0; ni < 2; ++ni) {
            const int col  = n0 + wn + ni * 16 + (lane & 15);
            const int rbse = m0 + wm + mi * 16 + ((lane >> 4) << 3);
#pragma unroll
            for (int r = 0; r < 8; ++r)
                oo[(size_t)(rbse + r) * NH + col] = acc[mi][ni][r];
        }
}

// ---------------------------------------------------------------------------
extern "C" void kernel_launch(void* const* d_in, const int* in_sizes, int n_in,
                              void* d_out, int out_size, void* d_ws, size_t ws_size,
                              hipStream_t stream) {
    const float* x  = (const float*)d_in[0];   // [E,T,H]
    const float* w1 = (const float*)d_in[1];   // [E,F,H]
    const float* w2 = (const float*)d_in[2];   // [E,F,H]
    float* out = (float*)d_out;                // [E,T,H]

    __bf16* a_ws = (__bf16*)d_ws;              // [E,T,F] bf16 = 128 MiB

    dim3 blk(256, 1, 1);
    dim3 g1(NF / BN, NT / BM, NE);             // 32 x 16 x 8
    moe_gemm1_gelu<<<g1, blk, 0, stream>>>(x, w1, a_ws);

    dim3 g2(NH / BN, NT / BM, NE);             // 8 x 16 x 8
    moe_gemm2<<<g2, blk, 0, stream>>>(a_ws, w2, out);
}
